// Context_ComplEx_63591285784884
// MI455X (gfx1250) — matrix operand order repaired and checked by reading, not compiled, not run on hardware
//
#include <hip/hip_runtime.h>
#include <hip/hip_bf16.h>

// ---------------------------------------------------------------------------
// Problem constants (from reference)
// ---------------------------------------------------------------------------
constexpr int N_ENT  = 100000;
constexpr int RANK   = 256;
constexpr int TWO_R  = 512;
constexpr int B_SZ   = 2048;
constexpr int MAX_NB = 50;

constexpr int NT_TOTAL = N_ENT / 16;   // 6250 N-tiles (exact)
constexpr int MT_TOTAL = B_SZ / 16;    // 128 M-tiles (exact)

// WMMA fragment / accumulator types (CDNA5 wave32)
typedef __attribute__((ext_vector_type(16))) __bf16 v16bf;
typedef __attribute__((ext_vector_type(8)))  float  v8f;

union Frag32 {            // 16 bf16 = 32 bytes = two uint4 loads
    uint4  u[2];
    v16bf  v;
};
static_assert(sizeof(v16bf) == 32, "v16bf must be 32B");

__device__ __forceinline__ unsigned short f2bf(float f) {
    // round-to-nearest-even f32 -> bf16 (inputs are finite)
    unsigned u = __float_as_uint(f);
    u += 0x7FFFu + ((u >> 16) & 1u);
    return (unsigned short)(u >> 16);
}

// ---------------------------------------------------------------------------
// Kernel 0: convert E_ent (f32, N_ENT x 512) -> bf16 copy in workspace
// ---------------------------------------------------------------------------
__global__ __launch_bounds__(256)
void cvt_bf16_kernel(const float2* __restrict__ src,
                     unsigned int* __restrict__ dst, long n2) {
    long i      = (long)blockIdx.x * blockDim.x + threadIdx.x;
    long stride = (long)gridDim.x * blockDim.x;
    for (; i < n2; i += stride) {
        float2 v = src[i];
        dst[i] = (unsigned)f2bf(v.x) | ((unsigned)f2bf(v.y) << 16);
    }
}

// ---------------------------------------------------------------------------
// Kernel 1: per-triple pipeline (one block of 256 threads per batch row)
//   gathers, w = trp@W + b, neighbor attention + softmax, ec, ec@W2 + b,
//   gate, gated_*, q_re/q_im (stored as bf16 into workspace), norms -> d_out
// ---------------------------------------------------------------------------
__global__ __launch_bounds__(256)
void stageA_kernel(const float* __restrict__ E_ent, const float* __restrict__ E_rel,
                   const float* __restrict__ W_re,  const float* __restrict__ W_im,
                   const float* __restrict__ b_w_re, const float* __restrict__ b_w_im,
                   const float* __restrict__ W2_re, const float* __restrict__ W2_im,
                   const float* __restrict__ b_w2_re, const float* __restrict__ b_w2_im,
                   const float* __restrict__ Wo_re, const float* __restrict__ Uo_re,
                   const float* __restrict__ Uo_im, const float* __restrict__ b_g,
                   const int* __restrict__ x, const int* __restrict__ nb_idx,
                   unsigned short* __restrict__ Qb,   // [B_SZ, 512] bf16
                   float* __restrict__ out_norms)     // d_out + B_SZ*N_ENT
{
    __shared__ float s_tre[TWO_R], s_tim[TWO_R];   // [lr|rr], [li|ri]
    __shared__ float s_wre[RANK],  s_wim[RANK];
    __shared__ float s_ecr[RANK],  s_eci[RANK];
    __shared__ float s_log[64];                    // logits -> alpha
    __shared__ float s_red[8];
    __shared__ float s_g, s_mx, s_sum;

    const int b    = blockIdx.x;
    const int t    = threadIdx.x;
    const int lane = t & 31;
    const int wave = t >> 5;

    const int hi = x[b * 3 + 0];
    const int re = x[b * 3 + 1];
    const int ti = x[b * 3 + 2];

    const float lr  = E_ent[(size_t)hi * TWO_R + t];
    const float li  = E_ent[(size_t)hi * TWO_R + RANK + t];
    const float rr  = E_rel[(size_t)re * TWO_R + t];
    const float rim = E_rel[(size_t)re * TWO_R + RANK + t];
    const float orr = E_ent[(size_t)ti * TWO_R + t];
    const float oii = E_ent[(size_t)ti * TWO_R + RANK + t];

    s_tre[t] = lr;  s_tre[RANK + t] = rr;
    s_tim[t] = li;  s_tim[RANK + t] = rim;

    const size_t nsz = (size_t)B_SZ * RANK;
    out_norms[(size_t)b * RANK + t]           = sqrtf(lr * lr + li * li);
    out_norms[nsz     + (size_t)b * RANK + t] = sqrtf(rr * rr + rim * rim);
    out_norms[2 * nsz + (size_t)b * RANK + t] = sqrtf(orr * orr + oii * oii);
    __syncthreads();

    // w = trp_re@W_re - trp_im@W_im + b ;  trp_re@W_im + trp_im@W_re + b
    float wre = b_w_re[t], wim = b_w_im[t];
    for (int k = 0; k < TWO_R; ++k) {
        const float tr = s_tre[k], tm = s_tim[k];
        const float wr = W_re[k * RANK + t], wi = W_im[k * RANK + t];
        wre += tr * wr - tm * wi;
        wim += tr * wi + tm * wr;
    }
    s_wre[t] = wre; s_wim[t] = wim;
    __syncthreads();

    // attention logits over 50 neighbors; waves stride over m
    for (int m = wave; m < MAX_NB; m += 8) {
        const size_t base = (size_t)nb_idx[b * MAX_NB + m] * TWO_R;
        float p = 0.f;
        #pragma unroll
        for (int j = 0; j < 8; ++j) {
            const int k = lane + 32 * j;
            p += s_wre[k] * E_ent[base + k] - s_wim[k] * E_ent[base + RANK + k];
        }
        for (int off = 16; off > 0; off >>= 1) p += __shfl_down(p, off, 32);
        if (lane == 0) s_log[m] = p;
    }
    __syncthreads();
    if (t == 0) {
        float mx = s_log[0];
        for (int m = 1; m < MAX_NB; ++m) mx = fmaxf(mx, s_log[m]);
        float sum = 0.f;
        for (int m = 0; m < MAX_NB; ++m) sum += __expf(s_log[m] - mx);
        s_mx = mx; s_sum = sum;
    }
    __syncthreads();
    if (t < MAX_NB) s_log[t] = __expf(s_log[t] - s_mx) / s_sum;   // alpha
    __syncthreads();

    // ec = alpha-weighted neighbor sum
    float ecr = 0.f, eci = 0.f;
    for (int m = 0; m < MAX_NB; ++m) {
        const size_t base = (size_t)nb_idx[b * MAX_NB + m] * TWO_R;
        const float a = s_log[m];
        ecr += a * E_ent[base + t];
        eci += a * E_ent[base + RANK + t];
    }
    s_ecr[t] = ecr; s_eci[t] = eci;
    __syncthreads();

    // ec2 = ec@W2 (complex) + b
    float e2r = b_w2_re[t], e2i = b_w2_im[t];
    for (int k = 0; k < RANK; ++k) {
        const float er = s_ecr[k], ei = s_eci[k];
        const float w2r = W2_re[k * RANK + t], w2i = W2_im[k * RANK + t];
        e2r += er * w2r - ei * w2i;
        e2i += er * w2i + ei * w2r;
    }

    // gate scalar
    float gp = (lr * rr - li * rim) * Uo_re[t]
             - (li * rr + lr * rim) * Uo_im[t]
             + e2r * Wo_re[t];
    for (int off = 16; off > 0; off >>= 1) gp += __shfl_down(gp, off, 32);
    if (lane == 0) s_red[wave] = gp;
    __syncthreads();
    if (t == 0) {
        float s = b_g[0];
        for (int wv = 0; wv < 8; ++wv) s += s_red[wv];
        s_g = 1.f / (1.f + __expf(-s));
    }
    __syncthreads();

    const float g   = s_g;
    const float gre = g * e2r + (1.f - g);
    const float gim = g * e2i;
    out_norms[3 * nsz + (size_t)b * RANK + t] = sqrtf(gre * gre + gim * gim);

    const float srrr = lr * rr, siri = li * rim, sirr = li * rr, srri = lr * rim;
    const float qre = (srrr + siri) * gre + (sirr + srri) * gim;
    const float qim = (srri + sirr) * gre + (siri - srrr) * gim;

    Qb[(size_t)b * TWO_R + t]        = f2bf(qre);
    Qb[(size_t)b * TWO_R + RANK + t] = f2bf(qim);
}

// ---------------------------------------------------------------------------
// Kernel 2: scores = [q_re|q_im] @ E_ent^T  (2048 x 100000, K=512) via
// v_wmma_f32_16x16x32_bf16. Per wave: 2 M-tiles x NT N-tiles of C.
// Templated on NT so the dominant NT==4 path has a branch-free inner loop
// and EXEC is provably all-ones at every WMMA (ISA requirement).
// ---------------------------------------------------------------------------
template<int NT>
__device__ __forceinline__
void gemm_body(const unsigned short* __restrict__ Qb,
               const unsigned short* __restrict__ Eb,
               float* __restrict__ scores,
               int mt0, int nt0, int lane)
{
    const int lane15 = lane & 15;

    // A-fragment (16x32 bf16, ISA layout): per lane two contiguous 8-K
    // segments at k0+ka and k0+16+ka, ka = (lane<16)?0:8, row M = lane&15.
    const size_t qrow0 = (size_t)(mt0 * 16 + lane15) * TWO_R;
    const size_t qrow1 = qrow0 + (size_t)16 * TWO_R;
    const int ka = (lane < 16) ? 0 : 8;
    // B-fragment (32x16 bf16): per lane two contiguous 8-K segments at
    // k0+kb and k0+16+kb, kb = (lane<16)?0:8, column N = lane&15.
    const int kb = (lane < 16) ? 0 : 8;

    size_t ebase[NT];
    #pragma unroll
    for (int n = 0; n < NT; ++n)
        ebase[n] = (size_t)((nt0 + n) * 16 + lane15) * TWO_R;

    v8f acc0[NT], acc1[NT];
    #pragma unroll
    for (int n = 0; n < NT; ++n) { acc0[n] = (v8f){}; acc1[n] = (v8f){}; }

    #pragma unroll 2
    for (int kc = 0; kc < TWO_R; kc += 32) {
        Frag32 a0, a1;
        a0.u[0] = *reinterpret_cast<const uint4*>(Qb + qrow0 + kc + ka);
        a0.u[1] = *reinterpret_cast<const uint4*>(Qb + qrow0 + kc + 16 + ka);
        a1.u[0] = *reinterpret_cast<const uint4*>(Qb + qrow1 + kc + ka);
        a1.u[1] = *reinterpret_cast<const uint4*>(Qb + qrow1 + kc + 16 + ka);
        #pragma unroll
        for (int n = 0; n < NT; ++n) {
            Frag32 bf;
            const uint4* bp = reinterpret_cast<const uint4*>(Eb + ebase[n] + kc + kb);
            bf.u[0] = bp[0];
            bf.u[1] = bp[2];
            acc0[n] = __builtin_amdgcn_wmma_f32_16x16x32_bf16(
                false, a0.v, false, bf.v, (short)0, acc0[n], false, false);
            acc1[n] = __builtin_amdgcn_wmma_f32_16x16x32_bf16(
                false, a1.v, false, bf.v, (short)0, acc1[n], false, false);
        }
    }

    // C layout: VGPR v holds M = v (+8 for lanes 16..31), N = lane&15.
    const int moff = (lane >> 4) << 3;
    #pragma unroll
    for (int n = 0; n < NT; ++n) {
        const int col = (nt0 + n) * 16 + lane15;
        const int r0  = mt0 * 16 + moff;
        const int r1  = r0 + 16;
        #pragma unroll
        for (int v = 0; v < 8; ++v) {
            __builtin_nontemporal_store(acc0[n][v], &scores[(size_t)(r0 + v) * N_ENT + col]);
            __builtin_nontemporal_store(acc1[n][v], &scores[(size_t)(r1 + v) * N_ENT + col]);
        }
    }
}

__global__ __launch_bounds__(256)
void scores_wmma_kernel(const unsigned short* __restrict__ Qb,  // [2048,512] bf16
                        const unsigned short* __restrict__ Eb,  // [100000,512] bf16
                        float* __restrict__ scores)             // [2048,100000] f32
{
    const int lane = threadIdx.x & 31;
    // Force the wave id into an SGPR so nt0/nvalid are compiler-provably
    // wave-uniform: the tile dispatch below becomes scalar branches and no
    // EXEC masking ever wraps a WMMA.
    const int wave = __builtin_amdgcn_readfirstlane(threadIdx.x >> 5);

    const int mt0 = blockIdx.y * 2;                    // 2 M-tiles per block
    const int nt0 = blockIdx.x * 32 + wave * 4;        // 4 N-tiles per wave
    int nvalid = NT_TOTAL - nt0;
    nvalid = nvalid < 0 ? 0 : (nvalid > 4 ? 4 : nvalid);

    switch (nvalid) {
    case 4: gemm_body<4>(Qb, Eb, scores, mt0, nt0, lane); break;  // 99.9% path
    case 3: gemm_body<3>(Qb, Eb, scores, mt0, nt0, lane); break;
    case 2: gemm_body<2>(Qb, Eb, scores, mt0, nt0, lane); break;
    case 1: gemm_body<1>(Qb, Eb, scores, mt0, nt0, lane); break;
    default: break;
    }
}

// ---------------------------------------------------------------------------
// Launch
// ---------------------------------------------------------------------------
extern "C" void kernel_launch(void* const* d_in, const int* in_sizes, int n_in,
                              void* d_out, int out_size, void* d_ws, size_t ws_size,
                              hipStream_t stream) {
    const float* E_ent   = (const float*)d_in[0];
    const float* E_rel   = (const float*)d_in[1];
    const float* W_re    = (const float*)d_in[2];
    const float* W_im    = (const float*)d_in[3];
    const float* b_w_re  = (const float*)d_in[4];
    const float* b_w_im  = (const float*)d_in[5];
    const float* W2_re   = (const float*)d_in[6];
    const float* W2_im   = (const float*)d_in[7];
    const float* b_w2_re = (const float*)d_in[8];
    const float* b_w2_im = (const float*)d_in[9];
    const float* Wo_re   = (const float*)d_in[10];
    // Wo_im (d_in[11]) is unused by the reference math (no q_im@Wo_im term in g)
    const float* Uo_re   = (const float*)d_in[12];
    const float* Uo_im   = (const float*)d_in[13];
    const float* b_g     = (const float*)d_in[14];
    const int*   x       = (const int*)d_in[15];
    const int*   nb_idx  = (const int*)d_in[16];

    float* scores = (float*)d_out;
    float* norms  = scores + (size_t)B_SZ * N_ENT;

    // Workspace: bf16 E_ent copy (102.4 MB) + bf16 Q (2 MB)
    unsigned short* Eb = (unsigned short*)d_ws;
    unsigned short* Qb = Eb + (size_t)N_ENT * TWO_R;

    // 0) E_ent -> bf16
    const long n2 = (long)N_ENT * TWO_R / 2;  // float2 count
    cvt_bf16_kernel<<<4096, 256, 0, stream>>>(
        (const float2*)E_ent, (unsigned int*)Eb, n2);

    // 1) per-triple pipeline -> Q (bf16) + norms
    stageA_kernel<<<B_SZ, 256, 0, stream>>>(
        E_ent, E_rel, W_re, W_im, b_w_re, b_w_im, W2_re, W2_im,
        b_w2_re, b_w2_im, Wo_re, Uo_re, Uo_im, b_g, x, nb_idx, Qb, norms);

    // 2) big GEMM on WMMA: 128 M-tiles (2/block) x 6250 N-tiles (32/block)
    dim3 grid((NT_TOTAL + 31) / 32, MT_TOTAL / 2);
    scores_wmma_kernel<<<grid, 256, 0, stream>>>(Qb, Eb, scores);
}
